// MultiHeadAttention_75161927680168
// MI455X (gfx1250) — matrix operand order repaired and checked by reading, not compiled, not run on hardware
//
#include <hip/hip_runtime.h>
#include <hip/hip_bf16.h>

// ---------------------------------------------------------------------------
// MI455X (gfx1250) MHA forward.  wave32, v_wmma_f32_16x16x32_bf16.
// All tensors that feed WMMA are stored in *fragment order* ([...][lane][16]
// bf16 = 32 B per lane), so operand loads are b128-wide (global or LDS)
// instead of 16x ds_load_u16.  Weight tiles are staged to LDS with
// global_load_async_to_lds_b128 (ASYNCcnt).
// ---------------------------------------------------------------------------

typedef __attribute__((ext_vector_type(16))) __bf16          v16bf;
typedef __attribute__((ext_vector_type(8)))  unsigned int    v8u;
typedef __attribute__((ext_vector_type(8)))  float           v8f;

#define D_MODEL 1024
#define N_HEAD  16
#define D_K     64
#define BATCH   4
#define SEQ     2048
#define M_ROWS  (BATCH * SEQ)   // 8192

__device__ __forceinline__ unsigned short f32_to_bf16(float f) {
  unsigned int u = __builtin_bit_cast(unsigned int, f);
  u += 0x7FFFu + ((u >> 16) & 1u);     // round-to-nearest-even
  return (unsigned short)(u >> 16);
}

// Load one 32-byte WMMA fragment (16 bf16) for this lane: 2x b128.
__device__ __forceinline__ v16bf frag_ld(const unsigned short* p) {
  v8u x = *(const v8u*)p;
  return __builtin_bit_cast(v16bf, x);
}

__device__ __forceinline__ v8f wmma_bf16(v16bf a, v16bf b, v8f c) {
  return __builtin_amdgcn_wmma_f32_16x16x32_bf16(false, a, false, b,
                                                 (short)0, c, false, false);
}

// CDNA5 async global->LDS copy (GV mode), tracked by ASYNCcnt.
__device__ __forceinline__ void async_copy_b128(unsigned lds_byte_off,
                                                const void* gsrc) {
  asm volatile("global_load_async_to_lds_b128 %0, %1, off"
               :: "v"(lds_byte_off), "v"(gsrc) : "memory");
}
__device__ __forceinline__ void wait_async0() {
  asm volatile("s_wait_asynccnt 0x0" ::: "memory");
}

// ---------------------------------------------------------------------------
// Fragment-order index helpers.
// A-fragment (16x32 bf16): lane = lo + 16*hi holds row M=lo,
//   K(i) = i<8 ? hi*8+i : 16+hi*8+(i-8)  ->  inverse: hi=(kk>>3)&1,
//   i=(kk&7)|((kk&16)>>1)
// B-fragment (32x16 bf16): lane = lo + 16*hi holds col N=lo, K(i) and inverse as for A.
// ---------------------------------------------------------------------------
__device__ __forceinline__ size_t qf_index(int bh, int s, int dk) {  // Q as A
  int qt = s >> 6, r = s & 63, w = r >> 4, lo = r & 15;
  int kh = dk >> 5, kk = dk & 31;
  int hi = (kk >> 3) & 1, i = (kk & 7) | ((kk & 16) >> 1);
  return ((((size_t)bh * 32 + qt) * 4 + w) * 2 + kh) * 512 +
         (size_t)(lo + 16 * hi) * 16 + i;
}
__device__ __forceinline__ size_t kf_index(int bh, int s, int dk) {  // K as B^T
  int jt = s >> 6, key = s & 63;                    // n = key, kdim = dk
  int t = key >> 4, lo = key & 15;
  int ks = dk >> 5, kk = dk & 31;
  int hi = (kk >> 3) & 1, i = (kk & 7) | ((kk & 16) >> 1);
  return ((((size_t)bh * 32 + jt) * 2 + ks) * 4 + t) * 512 +
         (size_t)(lo + 16 * hi) * 16 + i;
}
__device__ __forceinline__ size_t vf_index(int bh, int s, int dk) {  // V as B
  int jt = s >> 6, key = s & 63;                    // kdim = key, n = dk
  int ks = key >> 5, kk = key & 31;
  int hi = (kk >> 3) & 1, i = (kk & 7) | ((kk & 16) >> 1);
  int t = dk >> 4, lo = dk & 15;
  return ((((size_t)bh * 32 + jt) * 2 + ks) * 4 + t) * 512 +
         (size_t)(lo + 16 * hi) * 16 + i;
}
__device__ __forceinline__ size_t ctxf_index(int m, int n) {         // Ctx as A
  int mt = m >> 6, r = m & 63, w = r >> 4, lo = r & 15;
  int ks = n >> 5, kk = n & 31;
  int hi = (kk >> 3) & 1, i = (kk & 7) | ((kk & 16) >> 1);
  return (((size_t)mt * 4 + w) * 32 + ks) * 512 +
         (size_t)(lo + 16 * hi) * 16 + i;
}

// ---------------------------------------------------------------------------
// Kernel 0: pack a 1024x1024 f32 weight into bf16 B-fragment order:
//   Wp[((ct*32+ks)*4+t)*512 + lane*16 + i]
// ---------------------------------------------------------------------------
__global__ void pack_w_kernel(const float* __restrict__ W,
                              unsigned short* __restrict__ Wp) {
  int g = blockIdx.x * 256 + threadIdx.x;            // 1M elements
  int k = g >> 10, n = g & 1023;
  int ct = n >> 6, t = (n & 63) >> 4, lo = n & 15;
  int ks = k >> 5, kk = k & 31, hi = (kk >> 3) & 1, i = (kk & 7) | ((kk & 16) >> 1);
  size_t idx = (((size_t)ct * 32 + ks) * 4 + t) * 512 +
               (size_t)(lo + 16 * hi) * 16 + i;
  Wp[idx] = f32_to_bf16(W[(size_t)k * D_MODEL + n]);
}

// ---------------------------------------------------------------------------
// Kernel 1: projection GEMM.  X(f32 8192x1024) @ Wp(bf16 packed) + bias ->
// bf16 scattered into attention fragment layout (mode 0=Q/A, 1=K/B^T, 2=V/B).
// block 128 (4 waves); tile 64x64; k-step 32.
// ---------------------------------------------------------------------------
__global__ void proj_gemm_kernel(const float* __restrict__ X,
                                 const unsigned short* __restrict__ Wp,
                                 const float* __restrict__ bias,
                                 unsigned short* __restrict__ Out,
                                 int mode) {
  __shared__ __align__(16) unsigned short sAf[4][32][16];  // A frags, 4 KB
  __shared__ __align__(16) unsigned short sBf[4][32][16];  // B frags, 4 KB

  const int tid  = threadIdx.x;
  const int lane = tid & 31, w = tid >> 5;
  const int lo = lane & 15, hi = lane >> 4;
  const int row0 = blockIdx.x * 64, col0 = blockIdx.y * 64;
  const unsigned sb_base = (unsigned)(unsigned long long)(void*)&sBf[0][0][0];

  v8f acc[4];
#pragma unroll
  for (int t = 0; t < 4; ++t) acc[t] = (v8f){};

  for (int k0 = 0; k0 < D_MODEL; k0 += 32) {
    // --- stage B: contiguous 4 KB of packed weights via async-to-LDS ---
    const unsigned short* wp_tile =
        Wp + (((size_t)(col0 >> 6)) * 32 + (k0 >> 5)) * 2048;
    async_copy_b128(sb_base + tid * 16u,         wp_tile + (size_t)tid * 8);
    async_copy_b128(sb_base + (tid + 128) * 16u, wp_tile + (size_t)(tid + 128) * 8);

    // --- stage A: f32 quad load -> bf16 pack -> fragment-order b64 store ---
#pragma unroll
    for (int q = 0; q < 4; ++q) {
      int j  = tid + q * 128;                 // 512 quads
      int r  = j >> 3, qc = j & 7, kk = qc * 4;
      const float4 xv =
          *(const float4*)(X + (size_t)(row0 + r) * D_MODEL + k0 + kk);
      unsigned long long pk =
            (unsigned long long)f32_to_bf16(xv.x)
          | ((unsigned long long)f32_to_bf16(xv.y) << 16)
          | ((unsigned long long)f32_to_bf16(xv.z) << 32)
          | ((unsigned long long)f32_to_bf16(xv.w) << 48);
      int ahi = (kk >> 3) & 1;
      int ib  = (kk & 4) | ((kk & 16) >> 1);
      *(unsigned long long*)&sAf[r >> 4][(r & 15) + 16 * ahi][ib] = pk;
    }
    wait_async0();
    __syncthreads();

    v16bf a = frag_ld(&sAf[w][lane][0]);
#pragma unroll
    for (int t = 0; t < 4; ++t) {
      v16bf b = frag_ld(&sBf[t][lane][0]);
      acc[t] = wmma_bf16(a, b, acc[t]);
    }
    __syncthreads();
  }

  // Epilogue: +bias, bf16, scatter into the consumer's fragment layout.
#pragma unroll
  for (int t = 0; t < 4; ++t) {
    const int n = col0 + 16 * t + lo;
    const int h = n >> 6, dk = n & 63;
    const float bv = bias[n];
#pragma unroll
    for (int v = 0; v < 8; ++v) {
      const int m = row0 + 16 * w + v + hi * 8;
      const int b = m >> 11, s = m & (SEQ - 1);
      const int bh = b * N_HEAD + h;
      unsigned short val = f32_to_bf16(acc[t][v] + bv);
      size_t idx = (mode == 0) ? qf_index(bh, s, dk)
                 : (mode == 1) ? kf_index(bh, s, dk)
                               : vf_index(bh, s, dk);
      Out[idx] = val;
    }
  }
}

// ---------------------------------------------------------------------------
// Kernel 2: attention.  grid (B*H, S/64), block 128 (4 waves).
// Waves are independent 16-query strips -> no barriers.  Q/K/V fragments are
// loaded b128 straight from L2-resident global.  Pass 1: online max+sum.
// Pass 2: write attn once (DRAM roofline term) + P@V via LDS-staged P frags.
// ---------------------------------------------------------------------------
__global__ void attn_kernel(const unsigned short* __restrict__ Qf,
                            const unsigned short* __restrict__ Kf,
                            const unsigned short* __restrict__ Vf,
                            float* __restrict__ Attn,
                            unsigned short* __restrict__ Ctxf) {
  __shared__ __align__(16) unsigned short sPf[4][2][32][16];  // P frags, 8 KB
  __shared__ float rowmax[64];
  __shared__ float rowsum[64];

  const int tid  = threadIdx.x;
  const int lane = tid & 31, w = tid >> 5;
  const int lo = lane & 15, hi = lane >> 4;
  const int bh = blockIdx.x, qt = blockIdx.y;
  const int q0 = qt * 64, jend = q0 + 64;
  const float scale = 0.125f;                 // 1/sqrt(64)

  // Per-wave init of its own 16 rows (wave-private; no barrier needed).
  if (lane < 16) { rowmax[16 * w + lane] = -3.0e38f; rowsum[16 * w + lane] = 0.0f; }

  // Preload this wave's Q fragments (kept in VGPRs for the whole kernel).
  v16bf aq[2];
#pragma unroll
  for (int kh = 0; kh < 2; ++kh)
    aq[kh] = frag_ld(Qf + ((((size_t)bh * 32 + qt) * 4 + w) * 2 + kh) * 512 +
                     (size_t)lane * 16);

  // ---------------- pass 1: online row max + row sum(exp) ----------------
  for (int j0 = 0; j0 < jend; j0 += 64) {
    const int jt = j0 >> 6;
    v8f acc[4];
#pragma unroll
    for (int t = 0; t < 4; ++t) acc[t] = (v8f){};
#pragma unroll
    for (int kh = 0; kh < 2; ++kh) {
#pragma unroll
      for (int t = 0; t < 4; ++t) {
        v16bf bk = frag_ld(Kf + ((((size_t)bh * 32 + jt) * 2 + kh) * 4 + t) * 512 +
                           (size_t)lane * 16);
        acc[t] = wmma_bf16(aq[kh], bk, acc[t]);
      }
    }
#pragma unroll
    for (int v = 0; v < 8; ++v) {
      const int row = 16 * w + v + hi * 8;
      const int qg  = q0 + row;
      float sc[4];
      float tmax = -3.0e38f;
#pragma unroll
      for (int t = 0; t < 4; ++t) {
        sc[t] = acc[t][v] * scale;
        if (j0 + 16 * t + lo > qg) sc[t] = -1.0e9f;   // causal mask
        tmax = fmaxf(tmax, sc[t]);
      }
#pragma unroll
      for (int m = 1; m < 16; m <<= 1)
        tmax = fmaxf(tmax, __shfl_xor(tmax, m, 32));
      float part = 0.0f;
#pragma unroll
      for (int t = 0; t < 4; ++t) part += __expf(sc[t] - tmax);
#pragma unroll
      for (int m = 1; m < 16; m <<= 1)
        part += __shfl_xor(part, m, 32);
      if (lo == 0) {                                  // lanes 0 and 16
        float mo = rowmax[row];
        float mn = fmaxf(mo, tmax);
        rowsum[row] = rowsum[row] * __expf(mo - mn) + part * __expf(tmax - mn);
        rowmax[row] = mn;
      }
    }
  }

  // ---------------- pass 2: emit attn (write-once) + P@V ----------------
  v8f ctx[4];
#pragma unroll
  for (int t = 0; t < 4; ++t) ctx[t] = (v8f){};

  for (int j0 = 0; j0 < jend; j0 += 64) {
    const int jt = j0 >> 6;
    v8f acc[4];
#pragma unroll
    for (int t = 0; t < 4; ++t) acc[t] = (v8f){};
#pragma unroll
    for (int kh = 0; kh < 2; ++kh) {
#pragma unroll
      for (int t = 0; t < 4; ++t) {
        v16bf bk = frag_ld(Kf + ((((size_t)bh * 32 + jt) * 2 + kh) * 4 + t) * 512 +
                           (size_t)lane * 16);
        acc[t] = wmma_bf16(aq[kh], bk, acc[t]);
      }
    }
#pragma unroll
    for (int v = 0; v < 8; ++v) {
      const int rloc = v + hi * 8;                  // row within wave strip
      const int row  = 16 * w + rloc;
      const int qg   = q0 + row;
      const float rm  = rowmax[row];
      const float inv = 1.0f / rowsum[row];
#pragma unroll
      for (int t = 0; t < 4; ++t) {
        const int kg = j0 + 16 * t + lo;
        float sc = acc[t][v] * scale;
        if (kg > qg) sc = -1.0e9f;
        float p = __expf(sc - rm) * inv;
        Attn[((size_t)bh * SEQ + qg) * SEQ + kg] = p;
        // scatter p into A-fragment order (same-wave producer/consumer)
        const int key = 16 * t + lo;
        const int kh2 = key >> 5, kk = key & 31;
        const int ahi = (kk >> 3) & 1, ai = (kk & 7) | ((kk & 16) >> 1);
        sPf[w][kh2][rloc + 16 * ahi][ai] = f32_to_bf16(p);
      }
    }
#pragma unroll
    for (int kh = 0; kh < 2; ++kh) {
      v16bf ap = frag_ld(&sPf[w][kh][lane][0]);
#pragma unroll
      for (int t = 0; t < 4; ++t) {
        v16bf bv = frag_ld(Vf + ((((size_t)bh * 32 + jt) * 2 + kh) * 4 + t) * 512 +
                           (size_t)lane * 16);
        ctx[t] = wmma_bf16(ap, bv, ctx[t]);
      }
    }
  }

  // Zero-fill the fully-masked (future) key region of attn.
  {
    const size_t base = ((size_t)bh * SEQ + q0) * SEQ;
    const int rem = SEQ - jend;
    for (int i = tid; i < 64 * rem; i += 128) {
      int r = i / rem, c = i % rem;
      Attn[base + (size_t)r * SEQ + jend + c] = 0.0f;
    }
  }

  // Context -> A-fragment layout for the output GEMM.
  const int b = bh >> 4, h = bh & 15;
#pragma unroll
  for (int t = 0; t < 4; ++t) {
    const int n = h * D_K + 16 * t + lo;
#pragma unroll
    for (int v = 0; v < 8; ++v) {
      const int s = q0 + 16 * w + v + hi * 8;
      const int m = b * SEQ + s;
      Ctxf[ctxf_index(m, n)] = f32_to_bf16(ctx[t][v]);
    }
  }
}

// ---------------------------------------------------------------------------
// Kernel 3: output GEMM.  out(f32) = Ctxf(bf16, A-frag order) @ Wop + bo.
// A fragments come b128 straight from global (L2-resident); B via async LDS.
// ---------------------------------------------------------------------------
__global__ void out_gemm_kernel(const unsigned short* __restrict__ Ctxf,
                                const unsigned short* __restrict__ Wp,
                                const float* __restrict__ bias,
                                float* __restrict__ Out) {
  __shared__ __align__(16) unsigned short sBf[4][32][16];

  const int tid  = threadIdx.x;
  const int lane = tid & 31, w = tid >> 5;
  const int lo = lane & 15, hi = lane >> 4;
  const int row0 = blockIdx.x * 64, col0 = blockIdx.y * 64;
  const unsigned sb_base = (unsigned)(unsigned long long)(void*)&sBf[0][0][0];

  v8f acc[4];
#pragma unroll
  for (int t = 0; t < 4; ++t) acc[t] = (v8f){};

  for (int k0 = 0; k0 < D_MODEL; k0 += 32) {
    const unsigned short* wp_tile =
        Wp + (((size_t)(col0 >> 6)) * 32 + (k0 >> 5)) * 2048;
    async_copy_b128(sb_base + tid * 16u,         wp_tile + (size_t)tid * 8);
    async_copy_b128(sb_base + (tid + 128) * 16u, wp_tile + (size_t)(tid + 128) * 8);

    v16bf a = frag_ld(Ctxf + (((size_t)(row0 >> 6) * 4 + w) * 32 + (k0 >> 5)) * 512 +
                      (size_t)lane * 16);
    wait_async0();
    __syncthreads();

#pragma unroll
    for (int t = 0; t < 4; ++t) {
      v16bf b = frag_ld(&sBf[t][lane][0]);
      acc[t] = wmma_bf16(a, b, acc[t]);
    }
    __syncthreads();
  }

#pragma unroll
  for (int t = 0; t < 4; ++t) {
    const int n = col0 + 16 * t + lo;
    const float bv = bias[n];
#pragma unroll
    for (int v = 0; v < 8; ++v) {
      const int m = row0 + 16 * w + v + hi * 8;
      Out[(size_t)m * D_MODEL + n] = acc[t][v] + bv;
    }
  }
}

// ---------------------------------------------------------------------------
extern "C" void kernel_launch(void* const* d_in, const int* in_sizes, int n_in,
                              void* d_out, int out_size, void* d_ws, size_t ws_size,
                              hipStream_t stream) {
  (void)in_sizes; (void)n_in; (void)out_size; (void)ws_size;
  // Inputs (reference order): q,k,v,mask,Wq,bq,Wk,bk,Wv,bv,Wo,bo
  const float* k  = (const float*)d_in[1];   // reference projects Q from k (quirk)
  const float* v  = (const float*)d_in[2];
  const float* Wq = (const float*)d_in[4];
  const float* bq = (const float*)d_in[5];
  const float* Wk = (const float*)d_in[6];
  const float* bk = (const float*)d_in[7];
  const float* Wv = (const float*)d_in[8];
  const float* bv = (const float*)d_in[9];
  const float* Wo = (const float*)d_in[10];
  const float* bo = (const float*)d_in[11];

  // Workspace (bf16 elements): 4 packed weights (1M each) + Qf|Kf|Vf|Ctxf (8M each)
  unsigned short* Wqp  = (unsigned short*)d_ws;
  unsigned short* Wkp  = Wqp + (1u << 20);
  unsigned short* Wvp  = Wkp + (1u << 20);
  unsigned short* Wop  = Wvp + (1u << 20);
  unsigned short* Qf   = Wop + (1u << 20);
  unsigned short* Kf   = Qf  + ((size_t)1 << 23);
  unsigned short* Vf   = Kf  + ((size_t)1 << 23);
  unsigned short* Ctxf = Vf  + ((size_t)1 << 23);

  float* out  = (float*)d_out;                          // (B,S,D)
  float* attn = out + (size_t)M_ROWS * D_MODEL;         // (B,H,S,S)

  dim3 blk(128);
  pack_w_kernel<<<4096, 256, 0, stream>>>(Wq, Wqp);
  pack_w_kernel<<<4096, 256, 0, stream>>>(Wk, Wkp);
  pack_w_kernel<<<4096, 256, 0, stream>>>(Wv, Wvp);
  pack_w_kernel<<<4096, 256, 0, stream>>>(Wo, Wop);

  dim3 gGemm(M_ROWS / 64, D_MODEL / 64);                // 128 x 16
  proj_gemm_kernel<<<gGemm, blk, 0, stream>>>(k, Wqp, bq, Qf, 0);
  proj_gemm_kernel<<<gGemm, blk, 0, stream>>>(k, Wkp, bk, Kf, 1);
  proj_gemm_kernel<<<gGemm, blk, 0, stream>>>(v, Wvp, bv, Vf, 2);

  dim3 gAttn(BATCH * N_HEAD, SEQ / 64);                 // 64 x 32
  attn_kernel<<<gAttn, blk, 0, stream>>>(Qf, Kf, Vf, attn, Ctxf);

  out_gemm_kernel<<<gGemm, blk, 0, stream>>>(Ctxf, Wop, bo, out);
}
